// SwinTransformerBlock_4629974745663
// MI455X (gfx1250) — matrix operand order repaired and checked by reading, not compiled, not run on hardware
//
#include <hip/hip_runtime.h>
#include <cstdint>
#include <cstddef>

#define DEVI __device__ __forceinline__

typedef __attribute__((ext_vector_type(16))) __bf16 v16bf;
typedef __attribute__((ext_vector_type(8)))  float  v8f;

struct __align__(16) U128 { unsigned int x[4]; };
union FragU { v16bf v; U128 u[2]; };

// ---------------- problem constants ----------------
constexpr int C_    = 256;
constexpr int HID_  = 1024;
constexpr int IMGH  = 56, IMGW = 56;
constexpr int SH_   = 3;
constexpr int NWIN_ = 64;               // windows per image (8x8)
constexpr int WINTOT = 32 * NWIN_;      // 2048 windows total
constexpr int NRAW  = 49;               // tokens per window
constexpr int NPAD  = 64;               // padded tokens per window
constexpr int MPAD  = WINTOT * NPAD;    // 131072 padded rows
constexpr int TOK   = 32 * IMGH * IMGW; // 100352 real tokens

// ---------------- small helpers ----------------
DEVI unsigned short f2bf(float f) {
  unsigned u = __float_as_uint(f);
  if ((u & 0x7fffffffu) > 0x7f800000u) return (unsigned short)0x7fc0;
  unsigned r = u + 0x7fffu + ((u >> 16) & 1u);
  return (unsigned short)(r >> 16);
}

DEVI v8f vzero8() { v8f z;
#pragma unroll
  for (int i = 0; i < 8; ++i) z[i] = 0.f; return z; }

DEVI float gelu_exact(float x) {
  return 0.5f * x * (1.f + erff(x * 0.70710678118654752f));
}

// A fragment: 16x32 bf16, row-major source, per ISA layout:
// lane<16: M=lane, halves 0..7 = K[hi*8 .. +8), halves 8..15 = K[16+hi*8 .. +8)
DEVI v16bf load_frag_A(const unsigned short* Abase, int lda, int lane) {
  const unsigned short* pr = Abase + (size_t)(lane & 15) * lda + ((lane >> 4) << 3);
  FragU f;
  f.u[0] = *reinterpret_cast<const U128*>(pr);
  f.u[1] = *reinterpret_cast<const U128*>(pr + 16);
  return f.v;
}

// B fragment: 32x16 bf16 given B^T rows (n-major, k contiguous):
// lane reads B^T[n = lane&15][hi*8 .. +8) and [16+hi*8 .. +8), matching A.
DEVI v16bf load_frag_B(const unsigned short* Btbase, int ldb, int lane) {
  const unsigned short* pr = Btbase + (size_t)(lane & 15) * ldb + ((lane >> 4) << 3);
  FragU f;
  f.u[0] = *reinterpret_cast<const U128*>(pr);
  f.u[1] = *reinterpret_cast<const U128*>(pr + 16);
  return f.v;
}

DEVI v8f wmma_bf16(v16bf a, v16bf b, v8f c) {
  return __builtin_amdgcn_wmma_f32_16x16x32_bf16(false, a, false, b, (short)0, c, false, false);
}

// wave computes a 32x64 tile (2x4 WMMA tiles): 8 WMMAs per 6 fragment loads
DEVI void wave_mma_32x64(const unsigned short* A, int lda,
                         const unsigned short* Bt, int ldb,
                         int K, int lane, v8f acc[2][4]) {
  for (int k = 0; k < K; k += 32) {
    v16bf a0 = load_frag_A(A + k, lda, lane);
    v16bf a1 = load_frag_A(A + (size_t)16 * lda + k, lda, lane);
#pragma unroll
    for (int j = 0; j < 4; ++j) {
      v16bf bj = load_frag_B(Bt + (size_t)(16 * j) * ldb + k, ldb, lane);
      acc[0][j] = wmma_bf16(a0, bj, acc[0][j]);
      acc[1][j] = wmma_bf16(a1, bj, acc[1][j]);
    }
  }
}

// block-wide (256 thr) sum of two values
struct F2 { float a, b; };
DEVI F2 block_reduce2(float a, float b) {
  for (int off = 16; off > 0; off >>= 1) {
    a += __shfl_down(a, off, 32);
    b += __shfl_down(b, off, 32);
  }
  __shared__ float sa[8], sb[8];
  int wv = threadIdx.x >> 5, ln = threadIdx.x & 31;
  if (ln == 0) { sa[wv] = a; sb[wv] = b; }
  __syncthreads();
  if (wv == 0) {
    a = (ln < 8) ? sa[ln] : 0.f;
    b = (ln < 8) ? sb[ln] : 0.f;
    for (int off = 4; off > 0; off >>= 1) {
      a += __shfl_down(a, off, 32);
      b += __shfl_down(b, off, 32);
    }
    if (ln == 0) { sa[0] = a; sb[0] = b; }
  }
  __syncthreads();
  F2 r; r.a = sa[0]; r.b = sb[0]; return r;
}

// ---------------- kernels ----------------

__global__ void __launch_bounds__(256) k_f2bf(unsigned short* dst, const float* src, int n) {
  int i = blockIdx.x * 256 + threadIdx.x;
  if (i < n) dst[i] = f2bf(src[i]);
}

// LN1 + cyclic shift(-3,-3) + window partition, padded to 64 rows/window, bf16 out
__global__ void __launch_bounds__(256) k_ln_shift(const float* __restrict__ x,
                                                  const float* __restrict__ g,
                                                  const float* __restrict__ b,
                                                  unsigned short* __restrict__ Xw) {
  int row = blockIdx.x;        // [0, MPAD)
  int c   = threadIdx.x;       // [0, 256)
  int w = row >> 6, n = row & 63;
  if (n >= NRAW) { Xw[(size_t)row * C_ + c] = 0; return; }
  int bimg = w >> 6, wi = w & 63;
  int whi = wi >> 3, wwi = wi & 7;
  int r7 = n / 7, c7 = n % 7;
  int h0 = whi * 7 + r7 + SH_; if (h0 >= IMGH) h0 -= IMGH;
  int w0 = wwi * 7 + c7 + SH_; if (w0 >= IMGW) w0 -= IMGW;
  size_t src = ((size_t)bimg * (IMGH * IMGW) + h0 * IMGW + w0) * C_;
  float v = x[src + c];
  F2 s = block_reduce2(v, v * v);
  float mu  = s.a * (1.f / 256.f);
  float var = s.b * (1.f / 256.f) - mu * mu;
  float rs  = rsqrtf(var + 1e-5f);
  Xw[(size_t)row * C_ + c] = f2bf((v - mu) * rs * g[c] + b[c]);
}

// QKV GEMM: [MPAD,256] x W[768,256]^T, scatter to per-(window,head) Q/K/V^T bf16 tiles
__global__ void __launch_bounds__(256) k_gemm_qkv(const unsigned short* __restrict__ Xw,
                                                  const unsigned short* __restrict__ Wq,
                                                  const float* __restrict__ bias,
                                                  unsigned short* __restrict__ Qb,
                                                  unsigned short* __restrict__ Kb,
                                                  unsigned short* __restrict__ Vtb) {
  int lane = threadIdx.x & 31, wave = threadIdx.x >> 5;
  int m0 = blockIdx.x * 128 + (wave >> 1) * 32;
  int n0 = blockIdx.y * 128 + (wave & 1) * 64;
  v8f acc[2][4];
#pragma unroll
  for (int i = 0; i < 2; ++i) for (int j = 0; j < 4; ++j) acc[i][j] = vzero8();
  wave_mma_32x64(Xw + (size_t)m0 * C_, C_, Wq + (size_t)n0 * C_, C_, C_, lane, acc);
  int hi = lane >> 4, nl = lane & 15;
#pragma unroll
  for (int i = 0; i < 2; ++i)
#pragma unroll
    for (int j = 0; j < 4; ++j)
#pragma unroll
      for (int r = 0; r < 8; ++r) {
        int m = m0 + i * 16 + r + hi * 8;
        int n = n0 + j * 16 + nl;
        unsigned short bv = f2bf(acc[i][j][r] + bias[n]);
        int w = m >> 6, t = m & 63;
        int which = n >> 8, h = (n >> 5) & 7, d = n & 31;
        size_t base = (size_t)(w * 8 + h);
        if (which == 0)      Qb[(base * 64 + t) * 32 + d] = bv;
        else if (which == 1) Kb[(base * 64 + t) * 32 + d] = bv;
        else                 Vtb[(base * 32 + d) * 64 + t] = bv;
      }
}

// attention: 1 block per window (8 waves = 8 heads). S=QK^T (WMMA) -> LDS softmax+mask -> P@V (WMMA)
__global__ void __launch_bounds__(256) k_attn(const unsigned short* __restrict__ Qb,
                                              const unsigned short* __restrict__ Kb,
                                              const unsigned short* __restrict__ Vtb,
                                              const float* __restrict__ mask,
                                              unsigned short* __restrict__ Ow) {
  int lane = threadIdx.x & 31, h = threadIdx.x >> 5;
  int w = blockIdx.x, wi = w & 63;
  const unsigned short* Q  = Qb  + (size_t)(w * 8 + h) * 64 * 32;
  const unsigned short* Kp = Kb  + (size_t)(w * 8 + h) * 64 * 32;
  const unsigned short* Vt = Vtb + (size_t)(w * 8 + h) * 32 * 64;
  const float* mw = mask + (size_t)wi * NRAW * NRAW;

  __shared__ float Ssm[8][16][65];
  __shared__ __align__(16) unsigned short Pm[8][16][64];

  int hi = lane >> 4, nl = lane & 15;
  const float scale = 0.17677669529663687f;  // 1/sqrt(32)

  for (int mt = 0; mt < 4; ++mt) {
    v8f s[4];
#pragma unroll
    for (int i = 0; i < 4; ++i) s[i] = vzero8();
    v16bf a = load_frag_A(Q + mt * 16 * 32, 32, lane);
#pragma unroll
    for (int nt = 0; nt < 4; ++nt) {
      v16bf bk = load_frag_B(Kp + nt * 16 * 32, 32, lane);
      s[nt] = wmma_bf16(a, bk, s[nt]);
    }
    // stage scores to LDS with scale + shifted-window mask; kill padded key columns
#pragma unroll
    for (int nt = 0; nt < 4; ++nt)
#pragma unroll
      for (int r = 0; r < 8; ++r) {
        int lm = r + hi * 8;
        int mg = mt * 16 + lm;
        int ng = nt * 16 + nl;
        float v = s[nt][r] * scale;
        if (ng >= NRAW) v = -1e30f;
        else if (mg < NRAW) v += mw[mg * NRAW + ng];
        Ssm[h][lm][ng] = v;
      }
    __syncthreads();
    // softmax: lane nl owns row nl, column half [hi*32, hi*32+32)
    {
      float mx = -1e30f;
      for (int ci = 0; ci < 32; ++ci) mx = fmaxf(mx, Ssm[h][nl][hi * 32 + ci]);
      mx = fmaxf(mx, __shfl_xor(mx, 16, 32));
      float sum = 0.f;
      for (int ci = 0; ci < 32; ++ci) {
        float e = __expf(Ssm[h][nl][hi * 32 + ci] - mx);
        Ssm[h][nl][hi * 32 + ci] = e;
        sum += e;
      }
      sum += __shfl_xor(sum, 16, 32);
      float inv = 1.f / sum;
      for (int ci = 0; ci < 32; ++ci)
        Pm[h][nl][hi * 32 + ci] = f2bf(Ssm[h][nl][hi * 32 + ci] * inv);
    }
    __syncthreads();
    // O(16x32) = P(16x64) @ V(64x32); V^T rows are contiguous-over-token
    v8f o[2]; o[0] = vzero8(); o[1] = vzero8();
#pragma unroll
    for (int kt = 0; kt < 2; ++kt) {
      v16bf pa = load_frag_A(&Pm[h][0][0] + kt * 32, 64, lane);
#pragma unroll
      for (int nt2 = 0; nt2 < 2; ++nt2) {
        v16bf vb = load_frag_B(Vt + nt2 * 16 * 64 + kt * 32, 64, lane);
        o[nt2] = wmma_bf16(pa, vb, o[nt2]);
      }
    }
#pragma unroll
    for (int nt2 = 0; nt2 < 2; ++nt2)
#pragma unroll
      for (int r = 0; r < 8; ++r) {
        int mg  = mt * 16 + r + hi * 8;
        int col = h * 32 + nt2 * 16 + nl;
        Ow[((size_t)w * 64 + mg) * C_ + col] = f2bf(o[nt2][r]);
      }
    __syncthreads();
  }
}

// proj GEMM + window reverse + reverse shift -> fp32 h (attention residual dropped per reference)
__global__ void __launch_bounds__(256) k_gemm_proj(const unsigned short* __restrict__ Ow,
                                                   const unsigned short* __restrict__ Wp,
                                                   const float* __restrict__ bias,
                                                   float* __restrict__ Hres) {
  int lane = threadIdx.x & 31, wave = threadIdx.x >> 5;
  int m0 = blockIdx.x * 128 + (wave >> 1) * 32;
  int n0 = blockIdx.y * 128 + (wave & 1) * 64;
  v8f acc[2][4];
#pragma unroll
  for (int i = 0; i < 2; ++i) for (int j = 0; j < 4; ++j) acc[i][j] = vzero8();
  wave_mma_32x64(Ow + (size_t)m0 * C_, C_, Wp + (size_t)n0 * C_, C_, C_, lane, acc);
  int hi = lane >> 4, nl = lane & 15;
#pragma unroll
  for (int i = 0; i < 2; ++i)
#pragma unroll
    for (int j = 0; j < 4; ++j)
#pragma unroll
      for (int r = 0; r < 8; ++r) {
        int m = m0 + i * 16 + r + hi * 8;
        int n = n0 + j * 16 + nl;
        int w = m >> 6, t = m & 63;
        if (t >= NRAW) continue;
        int bimg = w >> 6, wi = w & 63;
        int whi = wi >> 3, wwi = wi & 7;
        int r7 = t / 7, c7 = t % 7;
        int hf = whi * 7 + r7 + SH_; if (hf >= IMGH) hf -= IMGH;
        int wf = wwi * 7 + c7 + SH_; if (wf >= IMGW) wf -= IMGW;
        Hres[((size_t)bimg * (IMGH * IMGW) + hf * IMGW + wf) * C_ + n] =
            acc[i][j][r] + bias[n];
      }
}

// LN2 over h -> bf16 for FC1
__global__ void __launch_bounds__(256) k_ln2(const float* __restrict__ Hres,
                                             const float* __restrict__ g,
                                             const float* __restrict__ b,
                                             unsigned short* __restrict__ Xm) {
  int tok = blockIdx.x, c = threadIdx.x;
  float v = Hres[(size_t)tok * C_ + c];
  F2 s = block_reduce2(v, v * v);
  float mu  = s.a * (1.f / 256.f);
  float var = s.b * (1.f / 256.f) - mu * mu;
  float rs  = rsqrtf(var + 1e-5f);
  Xm[(size_t)tok * C_ + c] = f2bf((v - mu) * rs * g[c] + b[c]);
}

// FC1 GEMM + exact GELU -> bf16
__global__ void __launch_bounds__(256) k_gemm_fc1(const unsigned short* __restrict__ Xm,
                                                  const unsigned short* __restrict__ W1,
                                                  const float* __restrict__ bias,
                                                  unsigned short* __restrict__ Y1) {
  int lane = threadIdx.x & 31, wave = threadIdx.x >> 5;
  int m0 = blockIdx.x * 128 + (wave >> 1) * 32;
  int n0 = blockIdx.y * 128 + (wave & 1) * 64;
  v8f acc[2][4];
#pragma unroll
  for (int i = 0; i < 2; ++i) for (int j = 0; j < 4; ++j) acc[i][j] = vzero8();
  wave_mma_32x64(Xm + (size_t)m0 * C_, C_, W1 + (size_t)n0 * C_, C_, C_, lane, acc);
  int hi = lane >> 4, nl = lane & 15;
#pragma unroll
  for (int i = 0; i < 2; ++i)
#pragma unroll
    for (int j = 0; j < 4; ++j)
#pragma unroll
      for (int r = 0; r < 8; ++r) {
        int m = m0 + i * 16 + r + hi * 8;
        int n = n0 + j * 16 + nl;
        Y1[(size_t)m * HID_ + n] = f2bf(gelu_exact(acc[i][j][r] + bias[n]));
      }
}

// FC2 GEMM + bias + residual(h) -> fp32 output
__global__ void __launch_bounds__(256) k_gemm_fc2(const unsigned short* __restrict__ Y1,
                                                  const unsigned short* __restrict__ W2,
                                                  const float* __restrict__ bias,
                                                  const float* __restrict__ Hres,
                                                  float* __restrict__ out) {
  int lane = threadIdx.x & 31, wave = threadIdx.x >> 5;
  int m0 = blockIdx.x * 128 + (wave >> 1) * 32;
  int n0 = blockIdx.y * 128 + (wave & 1) * 64;
  v8f acc[2][4];
#pragma unroll
  for (int i = 0; i < 2; ++i) for (int j = 0; j < 4; ++j) acc[i][j] = vzero8();
  wave_mma_32x64(Y1 + (size_t)m0 * HID_, HID_, W2 + (size_t)n0 * HID_, HID_, HID_, lane, acc);
  int hi = lane >> 4, nl = lane & 15;
#pragma unroll
  for (int i = 0; i < 2; ++i)
#pragma unroll
    for (int j = 0; j < 4; ++j)
#pragma unroll
      for (int r = 0; r < 8; ++r) {
        int m = m0 + i * 16 + r + hi * 8;
        int n = n0 + j * 16 + nl;
        size_t idx = (size_t)m * C_ + n;
        out[idx] = acc[i][j][r] + bias[n] + Hres[idx];
      }
}

// ---------------- launcher ----------------
extern "C" void kernel_launch(void* const* d_in, const int* in_sizes, int n_in,
                              void* d_out, int out_size, void* d_ws, size_t ws_size,
                              hipStream_t stream) {
  (void)in_sizes; (void)n_in; (void)out_size; (void)ws_size;
  const float* x     = (const float*)d_in[0];
  const float* amask = (const float*)d_in[1];
  const float* n1g   = (const float*)d_in[2];
  const float* n1b   = (const float*)d_in[3];
  const float* qkvw  = (const float*)d_in[4];
  const float* qkvb  = (const float*)d_in[5];
  const float* projw = (const float*)d_in[6];
  const float* projb = (const float*)d_in[7];
  const float* n2g   = (const float*)d_in[8];
  const float* n2b   = (const float*)d_in[9];
  const float* fc1w  = (const float*)d_in[10];
  const float* fc1b  = (const float*)d_in[11];
  const float* fc2w  = (const float*)d_in[12];
  const float* fc2b  = (const float*)d_in[13];
  float* out = (float*)d_out;

  char* p = (char*)d_ws;
  auto take = [&](size_t bytes) {
    char* r = p; p += (bytes + 255) & ~(size_t)255; return r;
  };
  unsigned short* Wqkv  = (unsigned short*)take((size_t)768 * 256 * 2);
  unsigned short* Wproj = (unsigned short*)take((size_t)256 * 256 * 2);
  unsigned short* Wfc1  = (unsigned short*)take((size_t)1024 * 256 * 2);
  unsigned short* Wfc2  = (unsigned short*)take((size_t)256 * 1024 * 2);
  unsigned short* Xw    = (unsigned short*)take((size_t)MPAD * C_ * 2);   // reused as Xm
  float*          Hres  = (float*)take((size_t)TOK * C_ * 4);
  char* regionA = p;
  unsigned short* Qb  = (unsigned short*)take((size_t)WINTOT * 8 * 64 * 32 * 2);
  unsigned short* Kb  = (unsigned short*)take((size_t)WINTOT * 8 * 64 * 32 * 2);
  unsigned short* Vtb = (unsigned short*)take((size_t)WINTOT * 8 * 32 * 64 * 2);
  unsigned short* Ow  = (unsigned short*)take((size_t)MPAD * C_ * 2);
  unsigned short* Y1  = (unsigned short*)regionA;  // overlaps Qb/Kb/Vtb (dead after proj)
  unsigned short* Xm  = Xw;                        // Xw dead after QKV GEMM

  // weight conversion fp32 -> bf16 (B^T == W row-major: exactly what the B-fragment wants)
  k_f2bf<<<(768 * 256 + 255) / 256, 256, 0, stream>>>(Wqkv, qkvw, 768 * 256);
  k_f2bf<<<(256 * 256 + 255) / 256, 256, 0, stream>>>(Wproj, projw, 256 * 256);
  k_f2bf<<<(1024 * 256 + 255) / 256, 256, 0, stream>>>(Wfc1, fc1w, 1024 * 256);
  k_f2bf<<<(256 * 1024 + 255) / 256, 256, 0, stream>>>(Wfc2, fc2w, 256 * 1024);

  k_ln_shift<<<MPAD, 256, 0, stream>>>(x, n1g, n1b, Xw);
  k_gemm_qkv<<<dim3(MPAD / 128, 768 / 128), 256, 0, stream>>>(Xw, Wqkv, qkvb, Qb, Kb, Vtb);
  k_attn<<<WINTOT, 256, 0, stream>>>(Qb, Kb, Vtb, amask, Ow);
  k_gemm_proj<<<dim3(MPAD / 128, 256 / 128), 256, 0, stream>>>(Ow, Wproj, projb, Hres);
  k_ln2<<<TOK, 256, 0, stream>>>(Hres, n2g, n2b, Xm);
  k_gemm_fc1<<<dim3(TOK / 128, 1024 / 128), 256, 0, stream>>>(Xm, Wfc1, fc1b, Y1);
  k_gemm_fc2<<<dim3(TOK / 128, 256 / 128), 256, 0, stream>>>(Y1, Wfc2, fc2b, Hres, out);
}
